// PoseHighResolutionNet_20641612825285
// MI455X (gfx1250) — compile-verified, matrix operand order checked by reading, not run on hardware
//
#include <hip/hip_runtime.h>
#include <hip/hip_bf16.h>
#include <cstddef>

// ---------------------------------------------------------------------------
// Shapes (from the reference)
// ---------------------------------------------------------------------------
#define NJ_   17
#define IC_   128
#define NB_   19
#define B_    2
#define H_    96
#define W_    72
#define HW_   (H_ * W_)          // 6912
#define NTOT_ (B_ * HW_)         // 13824 spatial positions
#define NBLK64_ (NTOT_ / 64)     // 216 N-tiles of 64
#define KSTEPS_ 36               // K = 128*9 = 1152 -> 36 chunks of 32

typedef __attribute__((ext_vector_type(16))) __bf16 v16bf;
typedef __attribute__((ext_vector_type(8)))  float  v8f;

__device__ inline unsigned short f2bf(float f) {
  union { __bf16 h; unsigned short u; } cv;
  cv.h = (__bf16)f;
  return cv.u;
}

// ---------------------------------------------------------------------------
// diff = y - x (fp32), zero helper
// ---------------------------------------------------------------------------
__global__ void k_diff(const float* __restrict__ x, const float* __restrict__ y,
                       float* __restrict__ d, int n) {
  int t = blockIdx.x * 256 + threadIdx.x;
  if (t < n) d[t] = y[t] - x[t];
}

__global__ void k_zero(float* __restrict__ p, int n) {
  int t = blockIdx.x * 256 + threadIdx.x;
  if (t < n) p[t] = 0.f;
}

// ---------------------------------------------------------------------------
// Weight repack: fp32 OIHW (Cin=128, 3x3) -> bf16 WMMA-A fragment order.
//   VGPR v = j/2, half p = j%2, lanegroup g = lane>>4
//   k = ((v&3)*2+p) + 8*g + 16*(v>>2) ; kglobal = step*32+k ; tap=kg/128, cin=kg%128
// ---------------------------------------------------------------------------
__global__ void k_pack(const float* __restrict__ w, unsigned short* __restrict__ ap,
                       int Cout, int CoutPad) {
  const int nt = CoutPad >> 4;
  const int total = KSTEPS_ * nt * 512;
  int t = blockIdx.x * 256 + threadIdx.x;
  if (t >= total) return;
  int e = t;
  const int j    = e & 15;  e >>= 4;
  const int lane = e & 31;  e >>= 5;
  const int mt   = e % nt;
  const int step = e / nt;
  const int m = mt * 16 + (lane & 15);
  const int g = lane >> 4;
  const int v = j >> 1, p = j & 1;
  const int k  = ((v & 3) * 2 + p) + 8 * g + 16 * (v >> 2);
  const int kg = step * 32 + k;
  const int tap = kg >> 7;       // / 128
  const int cin = kg & 127;
  float val = 0.f;
  if (m < Cout) val = w[((size_t)m * IC_ + cin) * 9 + tap];
  ap[t] = f2bf(val);
}

// ---------------------------------------------------------------------------
// WMMA implicit-GEMM 3x3 conv, Cin = 128 (NHWC bf16 input), pad = dilation.
// Block = 256 threads = 8 waves. Tile = 128(M) x 64(N).
// Double-buffered LDS B tiles: one barrier per k-step; the fill for step s+1
// (4 aligned global_load_b32 per thread) is issued before the 4 WMMAs of
// step s, hiding memory latency under matrix math. A-fragment pointer is
// advanced incrementally; next-tap spatial decode happens during the last
// k-chunk of the current tap. The trailing pipelined fill writes a dead
// buffer slot, so no guards are needed.
// ---------------------------------------------------------------------------
__global__ void k_wmma_conv(const unsigned short* __restrict__ inBf,   // [n][128] bf16
                            const unsigned short* __restrict__ apack,  // packed weights
                            const float* __restrict__ scale,           // [Cout] or null
                            const float* __restrict__ shift,
                            const float* __restrict__ resid,           // [n][Cout] or null
                            float* __restrict__ outF32,                // [n][Cout] or null
                            unsigned short* __restrict__ outBf,        // [n][128] or null
                            int Cout, int CoutPad, int dil, int relu) {
  __shared__ alignas(32) unsigned short ldsB[2 * 4 * 512];  // 2 buf x 4 sub-tiles

  const int t      = threadIdx.x;
  const int lane   = t & 31;
  const int wave   = t >> 5;
  const int nBase  = blockIdx.x * 64;
  const int ktiles = CoutPad >> 4;
  const int mt     = blockIdx.y * 8 + wave;
  const int mBase  = mt * 16;

  // --- fill-slot decode (thread t fills dword (fl,dw) of each sub-tile) ---
  const int fl = t >> 3;                   // fragment lane slot 0..31
  const int dw = t & 7;                    // dword within slot (VGPR index)
  const int fg = fl >> 4;
  const int fcol = fl & 15;
  const int kA = ((dw & 3) * 2) + 8 * fg + 16 * (dw >> 2);  // even k of dword

  int sb[4], syy[4], sxx[4];
#pragma unroll
  for (int s = 0; s < 4; ++s) {
    const int n   = nBase + s * 16 + fcol;
    sb[s]  = n / HW_;
    const int rem = n % HW_;
    syy[s] = rem / W_;
    sxx[s] = rem % W_;
  }

  const unsigned short* bp[4];
  unsigned int vmask[4];
  auto calcTap = [&](int tap_) {
    const int dy = (tap_ / 3 - 1) * dil;
    const int dx = (tap_ % 3 - 1) * dil;
#pragma unroll
    for (int s = 0; s < 4; ++s) {
      int ty = syy[s] + dy, tx = sxx[s] + dx;
      const bool ok = (ty >= 0) & (ty < H_) & (tx >= 0) & (tx < W_);
      vmask[s] = ok ? 0xFFFFFFFFu : 0u;
      int tyc = ty < 0 ? 0 : (ty > H_ - 1 ? H_ - 1 : ty);
      int txc = tx < 0 ? 0 : (tx > W_ - 1 ? W_ - 1 : tx);
      bp[s] = inBf + ((size_t)(sb[s] * HW_ + tyc * W_ + txc) * IC_);
    }
  };

  // A fragment pointer, advanced by one k-step each iteration
  const unsigned short* aptr = apack + ((size_t)mt * 32 + lane) * 16;
  const size_t astride = (size_t)ktiles * 512;

  v8f c0 = {}, c1 = {}, c2 = {}, c3 = {};

  // --- prologue: fill buffer 0 with step 0 ---
  calcTap(0);
  {
    unsigned int fv[4];
#pragma unroll
    for (int s = 0; s < 4; ++s)
      fv[s] = *reinterpret_cast<const unsigned int*>(bp[s] + kA) & vmask[s];
#pragma unroll
    for (int s = 0; s < 4; ++s)
      reinterpret_cast<unsigned int*>(ldsB)[s * 256 + t] = fv[s];
  }
  __syncthreads();

  for (int tap = 0; tap < 9; ++tap) {
#pragma unroll
    for (int cb = 0; cb < 4; ++cb) {
      const int p = cb & 1;                // buffer parity (tap*4 is even)
      // A fragment for this step (+ prefetch hint for the next one)
      v16bf a = *reinterpret_cast<const v16bf*>(aptr);
      __builtin_prefetch(aptr + astride, 0, 0);
      // next-step fill: issued before the WMMAs to hide latency
      if (cb == 3) calcTap(tap + 1 < 9 ? tap + 1 : 8);
      const int ncin0 = ((cb + 1) & 3) * 32;
      unsigned int fv[4];
#pragma unroll
      for (int s = 0; s < 4; ++s)
        fv[s] = *reinterpret_cast<const unsigned int*>(bp[s] + ncin0 + kA) & vmask[s];
      // consume current buffer
      v16bf b0 = *reinterpret_cast<const v16bf*>(&ldsB[p * 2048 + 0 * 512 + lane * 16]);
      v16bf b1 = *reinterpret_cast<const v16bf*>(&ldsB[p * 2048 + 1 * 512 + lane * 16]);
      v16bf b2 = *reinterpret_cast<const v16bf*>(&ldsB[p * 2048 + 2 * 512 + lane * 16]);
      v16bf b3 = *reinterpret_cast<const v16bf*>(&ldsB[p * 2048 + 3 * 512 + lane * 16]);
      c0 = __builtin_amdgcn_wmma_f32_16x16x32_bf16(false, a, false, b0, (short)0, c0, false, false);
      c1 = __builtin_amdgcn_wmma_f32_16x16x32_bf16(false, a, false, b1, (short)0, c1, false, false);
      c2 = __builtin_amdgcn_wmma_f32_16x16x32_bf16(false, a, false, b2, (short)0, c2, false, false);
      c3 = __builtin_amdgcn_wmma_f32_16x16x32_bf16(false, a, false, b3, (short)0, c3, false, false);
      // store next-step tile into the other buffer (dead store on last step)
#pragma unroll
      for (int s = 0; s < 4; ++s)
        reinterpret_cast<unsigned int*>(ldsB)[(p ^ 1) * 1024 + s * 256 + t] = fv[s];
      __syncthreads();
      aptr += astride;
    }
  }

  // --- epilogue: lane l covers N = nBase+s*16+(l&15), M = mBase+8*(l>>4)+r ---
  const int g    = lane >> 4;
  const int ecol = lane & 15;
  const int mb   = mBase + 8 * g;          // 8 contiguous output channels
  v8f accs[4] = {c0, c1, c2, c3};
#pragma unroll
  for (int s = 0; s < 4; ++s) {
    const int en = nBase + s * 16 + ecol;  // NHWC: linear n, no decode needed
    float v[8];
#pragma unroll
    for (int r = 0; r < 8; ++r) {
      float vv = accs[s][r];
      const int m = mb + r;
      if (scale) vv = vv * scale[m] + shift[m];
      v[r] = vv;
    }
    if (resid) {
#pragma unroll
      for (int r = 0; r < 8; ++r) v[r] += resid[(size_t)en * Cout + mb + r];
    }
    if (relu) {
#pragma unroll
      for (int r = 0; r < 8; ++r) v[r] = v[r] > 0.f ? v[r] : 0.f;
    }
    if (outF32) {
      if (mb + 7 < Cout) {
#pragma unroll
        for (int r = 0; r < 8; ++r) outF32[(size_t)en * Cout + mb + r] = v[r];
      } else {
#pragma unroll
        for (int r = 0; r < 8; ++r)
          if (mb + r < Cout) outF32[(size_t)en * Cout + mb + r] = v[r];
      }
    }
    if (outBf) {                            // only used when Cout == 128
      uint4 q;
      q.x = (unsigned int)f2bf(v[0]) | ((unsigned int)f2bf(v[1]) << 16);
      q.y = (unsigned int)f2bf(v[2]) | ((unsigned int)f2bf(v[3]) << 16);
      q.z = (unsigned int)f2bf(v[4]) | ((unsigned int)f2bf(v[5]) << 16);
      q.w = (unsigned int)f2bf(v[6]) | ((unsigned int)f2bf(v[7]) << 16);
      *reinterpret_cast<uint4*>(outBf + (size_t)en * IC_ + mb) = q;
    }
  }
}

// ---------------------------------------------------------------------------
// Direct fp32 conv (stem: 17->128 3x3 and 17->128 1x1). Negligible FLOPs.
// Reads NCHW fp32 diff, writes NHWC fp32 / bf16.
// ---------------------------------------------------------------------------
__global__ void k_dconv(const float* __restrict__ in, const float* __restrict__ w,
                        const float* __restrict__ scale, const float* __restrict__ shift,
                        float* __restrict__ outF32, unsigned short* __restrict__ outBf,
                        int Cin, int Cout, int ksz, int pad, int relu) {
  const int total = B_ * Cout * HW_;
  int t = blockIdx.x * 256 + threadIdx.x;
  if (t >= total) return;
  int xx = t % W_; int e = t / W_;
  int yy = e % H_; e /= H_;
  int m  = e % Cout;
  int b  = e / Cout;
  float acc = 0.f;
  for (int cidx = 0; cidx < Cin; ++cidx)
    for (int ky = 0; ky < ksz; ++ky)
      for (int kx = 0; kx < ksz; ++kx) {
        int ty = yy + ky - pad, tx = xx + kx - pad;
        if (ty >= 0 && ty < H_ && tx >= 0 && tx < W_)
          acc += w[(((size_t)m * Cin + cidx) * ksz + ky) * ksz + kx] *
                 in[(((size_t)b * Cin + cidx) * H_ + ty) * W_ + tx];
      }
  float v = acc * scale[m] + shift[m];
  if (relu) v = v > 0.f ? v : 0.f;
  const size_t on = (size_t)b * HW_ + yy * W_ + xx;   // NHWC
  if (outF32) outF32[on * Cout + m] = v;
  if (outBf)  outBf[on * IC_ + m]   = f2bf(v);
}

// ---------------------------------------------------------------------------
// Deformable conv (17->17, bilinear gather) accumulating 0.2 * result.
// Offsets are NHWC fp32 [n][306]; x and out stay NCHW fp32.
// ---------------------------------------------------------------------------
__global__ void k_deform(const float* __restrict__ x, const float* __restrict__ off,
                         const float* __restrict__ w, float* __restrict__ out, int dil) {
  int t = blockIdx.x * 256 + threadIdx.x;
  if (t >= NTOT_) return;
  int xx = t % W_; int e = t / W_;
  int yy = e % H_;
  int b  = e / H_;
  const size_t n306 = ((size_t)b * HW_ + yy * W_ + xx) * 306;
  float acc[NJ_];
#pragma unroll
  for (int o = 0; o < NJ_; ++o) acc[o] = 0.f;
  for (int cidx = 0; cidx < NJ_; ++cidx) {
    const float* img = x + ((size_t)b * NJ_ + cidx) * HW_;
    for (int k = 0; k < 9; ++k) {
      const int ch = (cidx * 9 + k) * 2;
      float dy = off[n306 + ch];
      float dx = off[n306 + ch + 1];
      float py = dy + (float)((k / 3 - 1) * dil) + (float)yy;
      float px = dx + (float)((k % 3 - 1) * dil) + (float)xx;
      float fy = floorf(py), fx = floorf(px);
      float wy = py - fy, wx = px - fx;
      int y0 = (int)fy, x0 = (int)fx;
      float vs[4];
#pragma unroll
      for (int q = 0; q < 4; ++q) {
        int yi = y0 + (q >> 1);
        int xi = x0 + (q & 1);
        int yc = yi < 0 ? 0 : (yi > H_ - 1 ? H_ - 1 : yi);
        int xc = xi < 0 ? 0 : (xi > W_ - 1 ? W_ - 1 : xi);
        float vv = img[yc * W_ + xc];
        vs[q] = (yi >= 0 && yi < H_ && xi >= 0 && xi < W_) ? vv : 0.f;
      }
      float s = vs[0] * (1 - wy) * (1 - wx) + vs[1] * (1 - wy) * wx +
                vs[2] * wy * (1 - wx)       + vs[3] * wy * wx;
#pragma unroll
      for (int o = 0; o < NJ_; ++o)
        acc[o] += w[((size_t)o * NJ_ + cidx) * 9 + k] * s;
    }
  }
  for (int o = 0; o < NJ_; ++o) {
    size_t oidx = (((size_t)b * NJ_ + o) * H_ + yy) * W_ + xx;
    out[oidx] += 0.2f * acc[o];
  }
}

// ---------------------------------------------------------------------------
// Host-side orchestration
// ---------------------------------------------------------------------------
extern "C" void kernel_launch(void* const* d_in, const int* in_sizes, int n_in,
                              void* d_out, int out_size, void* d_ws, size_t ws_size,
                              hipStream_t stream) {
  const float* x       = (const float*)d_in[0];
  const float* y       = (const float*)d_in[1];
  const float* blk0_w1 = (const float*)d_in[2];
  const float* blk0_s1 = (const float*)d_in[3];
  const float* blk0_b1 = (const float*)d_in[4];
  const float* blk0_w2 = (const float*)d_in[5];
  const float* blk0_s2 = (const float*)d_in[6];
  const float* blk0_b2 = (const float*)d_in[7];
  const float* blk0_wd = (const float*)d_in[8];
  const float* blk0_sd = (const float*)d_in[9];
  const float* blk0_bd = (const float*)d_in[10];
  const float* blk_w1  = (const float*)d_in[11];
  const float* blk_s1  = (const float*)d_in[12];
  const float* blk_b1  = (const float*)d_in[13];
  const float* blk_w2  = (const float*)d_in[14];
  const float* blk_s2  = (const float*)d_in[15];
  const float* blk_b2  = (const float*)d_in[16];
  const float* off_w   = (const float*)d_in[17];
  const float* dc_w    = (const float*)d_in[18];
  float* out = (float*)d_out;

  // --- workspace carve-out (256B aligned) ---
  char* ws = (char*)d_ws;
  size_t cur = 0;
  auto carve = [&](size_t bytes) -> void* {
    cur = (cur + 255) & ~(size_t)255;
    void* p = ws + cur;
    cur += bytes;
    return p;
  };
  const size_t actF32 = (size_t)NTOT_ * IC_ * sizeof(float);
  const size_t actBF  = (size_t)NTOT_ * IC_ * sizeof(unsigned short);
  const size_t pw128  = (size_t)KSTEPS_ * 8  * 512 * sizeof(unsigned short);
  const size_t pw384  = (size_t)KSTEPS_ * 24 * 512 * sizeof(unsigned short);

  float*          diff  = (float*)carve((size_t)B_ * NJ_ * HW_ * sizeof(float));
  float*          hF32A = (float*)carve(actF32);
  float*          hF32B = (float*)carve(actF32);
  unsigned short* hBfA  = (unsigned short*)carve(actBF);
  unsigned short* hBfB  = (unsigned short*)carve(actBF);
  unsigned short* tBf   = (unsigned short*)carve(actBF);
  unsigned short* pw0   = (unsigned short*)carve(pw128);
  unsigned short* pwB1  = (unsigned short*)carve(pw128 * NB_);
  unsigned short* pwB2  = (unsigned short*)carve(pw128 * NB_);
  unsigned short* pwOff = (unsigned short*)carve(pw384 * 5);
  float*          offB  = (float*)carve((size_t)NTOT_ * 306 * sizeof(float));
  (void)ws_size; (void)in_sizes; (void)n_in;

  const int pw128elems = KSTEPS_ * 8 * 512;   // 147456
  const int pw384elems = KSTEPS_ * 24 * 512;  // 442368

  // --- 1. repack all WMMA-path weights (fp32 -> bf16 fragment order) ---
  k_pack<<<dim3((pw128elems + 255) / 256), dim3(256), 0, stream>>>(blk0_w2, pw0, IC_, IC_);
  for (int l = 0; l < NB_; ++l) {
    k_pack<<<dim3((pw128elems + 255) / 256), dim3(256), 0, stream>>>(
        blk_w1 + (size_t)l * IC_ * IC_ * 9, pwB1 + (size_t)l * pw128elems, IC_, IC_);
    k_pack<<<dim3((pw128elems + 255) / 256), dim3(256), 0, stream>>>(
        blk_w2 + (size_t)l * IC_ * IC_ * 9, pwB2 + (size_t)l * pw128elems, IC_, IC_);
  }
  for (int i = 0; i < 5; ++i)
    k_pack<<<dim3((pw384elems + 255) / 256), dim3(256), 0, stream>>>(
        off_w + (size_t)i * 306 * IC_ * 9, pwOff + (size_t)i * pw384elems, 306, 384);

  // --- 2. diff = y - x ---
  const int ndiff = B_ * NJ_ * HW_;
  k_diff<<<dim3((ndiff + 255) / 256), dim3(256), 0, stream>>>(x, y, diff, ndiff);

  // --- 3. stem (direct fp32: cheap) ---
  k_dconv<<<dim3((B_ * IC_ * HW_ + 255) / 256), dim3(256), 0, stream>>>(
      diff, blk0_wd, blk0_sd, blk0_bd, hF32B, (unsigned short*)nullptr,
      NJ_, IC_, 1, 0, 0);
  k_dconv<<<dim3((B_ * IC_ * HW_ + 255) / 256), dim3(256), 0, stream>>>(
      diff, blk0_w1, blk0_s1, blk0_b1, (float*)nullptr, tBf,
      NJ_, IC_, 3, 1, 1);
  k_wmma_conv<<<dim3(NBLK64_, 1), dim3(256), 0, stream>>>(
      tBf, pw0, blk0_s2, blk0_b2, hF32B, hF32A, hBfA, IC_, IC_, 1, 1);

  // --- 4. 19 basic blocks (all WMMA) ---
  int curbuf = 0;
  for (int l = 0; l < NB_; ++l) {
    float*          hF = curbuf == 0 ? hF32A : hF32B;
    unsigned short* hB = curbuf == 0 ? hBfA  : hBfB;
    float*          oF = curbuf == 0 ? hF32B : hF32A;
    unsigned short* oB = curbuf == 0 ? hBfB  : hBfA;
    k_wmma_conv<<<dim3(NBLK64_, 1), dim3(256), 0, stream>>>(
        hB, pwB1 + (size_t)l * pw128elems, blk_s1 + l * IC_, blk_b1 + l * IC_,
        (const float*)nullptr, (float*)nullptr, tBf, IC_, IC_, 1, 1);
    k_wmma_conv<<<dim3(NBLK64_, 1), dim3(256), 0, stream>>>(
        tBf, pwB2 + (size_t)l * pw128elems, blk_s2 + l * IC_, blk_b2 + l * IC_,
        hF, oF, oB, IC_, IC_, 1, 1);
    curbuf ^= 1;
  }
  unsigned short* featBf = curbuf == 0 ? hBfA : hBfB;

  // --- 5. offset convs + deformable warps, accumulating 0.2*sum into d_out ---
  k_zero<<<dim3((out_size + 255) / 256), dim3(256), 0, stream>>>(out, out_size);
  static const int DILS[5] = {3, 6, 12, 18, 24};
  for (int i = 0; i < 5; ++i) {
    const int dil = DILS[i];
    k_wmma_conv<<<dim3(NBLK64_, 3), dim3(256), 0, stream>>>(
        featBf, pwOff + (size_t)i * pw384elems,
        (const float*)nullptr, (const float*)nullptr, (const float*)nullptr,
        offB, (unsigned short*)nullptr, 306, 384, dil, 0);
    k_deform<<<dim3((NTOT_ + 255) / 256), dim3(256), 0, stream>>>(
        x, offB, dc_w + (size_t)i * NJ_ * NJ_ * 9, out, dil);
  }
}
